// HSTGNN_63393717289326
// MI455X (gfx1250) — matrix operand order repaired and checked by reading, not compiled, not run on hardware
//
#include <hip/hip_runtime.h>
#include <hip/hip_bf16.h>

typedef _Float16 half_t;
typedef __attribute__((ext_vector_type(16))) _Float16 v16h;
typedef __attribute__((ext_vector_type(8)))  float    v8f;

#define DEV_INLINE __device__ __forceinline__

// ---- problem dimensions -------------------------------------------------
#define BB   32      // batch
#define TT   336     // time steps
#define NN   512     // nodes
#define DD   256     // feature dim D
#define MM   848     // N + T
#define MMP  864     // K-pad of 848 (27*32)
#define F1   352     // T + DS (already 11*32)
#define F2P  544     // pad of N+DS+DT=532 (17*32)
#define G1   768     // 3*D
#define G2   1536    // 3*N
#define TP   352     // pad of T=336 for rs GEMM K
#define KMAX 864     // largest K used by any GEMM (LDS A-tile bound)

DEV_INLINE float eluf(float x)      { return x > 0.f ? x : (expf(x) - 1.f); }
DEV_INLINE float sigmoidf_(float x) { return 1.f / (1.f + expf(-x)); }

// ---- WMMA fragment loaders (ISA 7.12.2 layouts, 16-bit A 16x32 / B 32x16)
DEV_INLINE v16h frag_a(const half_t* __restrict__ base, int ld, int lane, int kc) {
    const int m = lane & 15, hf = lane >> 4;
    const half_t* p = base + (long long)m * ld + kc;
    v16h a;
#pragma unroll
    for (int i = 0; i < 8; ++i) {
        const int kk = (i < 4) ? (2 * i + 8 * hf) : (16 + 2 * (i - 4) + 8 * hf);
        a[2 * i]     = p[kk];
        a[2 * i + 1] = p[kk + 1];
    }
    return a;
}
DEV_INLINE v16h frag_b(const half_t* __restrict__ base, int ld, int lane, int kc) {
    const int n = lane & 15, hf = lane >> 4;
    const half_t* p = base + (long long)n * ld + kc;
    v16h b;
#pragma unroll
    for (int i = 0; i < 8; ++i) {
        const int kk = (i < 4) ? (2 * i + 8 * hf) : (16 + 2 * (i - 4) + 8 * hf);
        b[2 * i]     = p[kk];
        b[2 * i + 1] = p[kk + 1];
    }
    return b;
}

// ---- generic batched GEMM: C = act(A * B^T + bias), optional f32/f16/f16^T outs
struct GemmP {
    const half_t* A;  long long sA;  int lda;
    const half_t* Bt; long long sB;  int ldb;
    const float*  bias;
    float*  C32;  long long sC;   int ldc;
    half_t* Ch;   long long sCh;  int ldch;
    half_t* ChT;  long long sChT; int ldchT;
    int M, N, K;        // K multiple of 32, M/N multiples of 16
    int act;            // 0=none, 1=elu, 2=tanh(relu(x))
    int epsDiag;        // add 0.1 on (row==col)
};

DEV_INLINE void gemm_epilogue(const GemmP& p, int bz, int rt, int ct, int lane,
                              const v8f& c) {
    const int n = lane & 15, hf = lane >> 4;
    const int col = ct * 16 + n;
    const float bias = p.bias ? p.bias[col] : 0.f;
#pragma unroll
    for (int r = 0; r < 8; ++r) {
        const int row = rt * 16 + r + 8 * hf;
        float v = c[r] + bias;
        if (p.act == 1)      v = eluf(v);
        else if (p.act == 2) v = (v <= 0.f) ? 0.f : tanhf(v);
        if (p.epsDiag && row == col) v += 0.1f;
        if (p.C32) p.C32[(long long)bz * p.sC   + (long long)row * p.ldc   + col] = v;
        if (p.Ch)  p.Ch [(long long)bz * p.sCh  + (long long)row * p.ldch  + col] = (half_t)v;
        if (p.ChT) p.ChT[(long long)bz * p.sChT + (long long)col * p.ldchT + row] = (half_t)v;
    }
}

// 8 waves/block; block owns one 16-row tile and 16 column tiles (2 per wave).
// A tile (16 x K) is staged once in LDS and shared by all waves.
__global__ __launch_bounds__(256) void k_gemm(GemmP p) {
    __shared__ half_t Atile[16 * KMAX];
    const int tid = threadIdx.x;
    const int wave = tid >> 5, lane = tid & 31;
    const int rt = blockIdx.y, bz = blockIdx.z;
    const int ntiles = p.N >> 4;
    const int khalf = p.K >> 1;            // dwords per A row

    // cooperative A-tile stage (dword granularity; lda,K are even)
    const half_t* Ag = p.A + (long long)bz * p.sA + (long long)rt * 16 * p.lda;
    for (int i = tid; i < 16 * khalf; i += 256) {
        const int r = i / khalf, c = i - r * khalf;
        ((unsigned int*)Atile)[i] =
            *(const unsigned int*)(Ag + (long long)r * p.lda + 2 * c);
    }
    __syncthreads();

    int ct0 = (blockIdx.x * 8 + wave) * 2;
    int ct1 = ct0 + 1;
    if (ct0 > ntiles - 1) ct0 = ntiles - 1;   // clamp: dup work, never OOB
    if (ct1 > ntiles - 1) ct1 = ntiles - 1;
    const half_t* B0 = p.Bt + (long long)bz * p.sB + (long long)ct0 * 16 * p.ldb;
    const half_t* B1 = p.Bt + (long long)bz * p.sB + (long long)ct1 * 16 * p.ldb;

    v8f c0 = {}, c1 = {};
    for (int kc = 0; kc < p.K; kc += 32) {
        v16h a  = frag_a(Atile, p.K, lane, kc);
        v16h b0 = frag_b(B0, p.ldb, lane, kc);
        v16h b1 = frag_b(B1, p.ldb, lane, kc);
        c0 = __builtin_amdgcn_wmma_f32_16x16x32_f16(false, a, false, b0,
                                                    (short)0, c0, false, false);
        c1 = __builtin_amdgcn_wmma_f32_16x16x32_f16(false, a, false, b1,
                                                    (short)0, c1, false, false);
    }
    gemm_epilogue(p, bz, rt, ct0, lane, c0);
    gemm_epilogue(p, bz, rt, ct1, lane, c1);
}

// ---- GRU scans: 16 batch rows per block, h kept in LDS (f32 + f16 copies)
__global__ __launch_bounds__(256) void k_gru1(const float* __restrict__ gi,
                                              const half_t* __restrict__ Whht,
                                              const float* __restrict__ bhh,
                                              half_t* __restrict__ outH,
                                              half_t* __restrict__ outHT) {
    extern __shared__ char smem[];
    float*  h32 = (float*)smem;               // 16*256
    float*  gh  = h32 + 16 * DD;              // 16*768
    half_t* h16 = (half_t*)(gh + 16 * G1);    // 16*256
    const int tid = threadIdx.x, wave = tid >> 5, lane = tid & 31;
    const int bb0 = blockIdx.z * 16;
    for (int i = tid; i < 16 * DD; i += 256) { h32[i] = 0.f; h16[i] = (half_t)0.f; }
    __syncthreads();

    for (int t = 0; t < TT; ++t) {
        // gh = h * Whh^T   (16 x 768, K = 256)
        for (int tile = wave; tile < G1 / 16; tile += 8) {
            v8f c = {};
            const half_t* Bt = Whht + tile * 16 * DD;
            for (int kc = 0; kc < DD; kc += 32) {
                v16h a = frag_a(h16, DD, lane, kc);
                v16h b = frag_b(Bt, DD, lane, kc);
                c = __builtin_amdgcn_wmma_f32_16x16x32_f16(false, a, false, b,
                                                           (short)0, c, false, false);
            }
            const int n = lane & 15, hf = lane >> 4;
#pragma unroll
            for (int r = 0; r < 8; ++r)
                gh[(r + 8 * hf) * G1 + tile * 16 + n] = c[r];
        }
        __syncthreads();
        // gates (torch order r,z,n)
        for (int i = tid; i < 16 * DD; i += 256) {
            const int b = i >> 8, d = i & 255;
            const float* g = gi + ((long long)(bb0 + b) * TT + t) * G1;
            const float rr = sigmoidf_(g[d]       + gh[b * G1 + d]       + bhh[d]);
            const float zz = sigmoidf_(g[DD + d]  + gh[b * G1 + DD + d]  + bhh[DD + d]);
            const float nn = tanhf(g[2 * DD + d] + rr * (gh[b * G1 + 2 * DD + d] + bhh[2 * DD + d]));
            const float h = (1.f - zz) * nn + zz * h32[i];
            h32[i] = h; h16[i] = (half_t)h;
            const long long bb = bb0 + b;
            outH [(bb * MM + NN + t) * (long long)DD + d]  = (half_t)h;
            outHT[(bb * DD + d) * (long long)MMP + NN + t] = (half_t)h;
        }
        __syncthreads();
    }
}

__global__ __launch_bounds__(256) void k_gru2(const float* __restrict__ gi,
                                              const half_t* __restrict__ Whht,
                                              const float* __restrict__ bhh,
                                              float* __restrict__ out) {
    extern __shared__ char smem[];
    float*  h32 = (float*)smem;               // 16*512
    float*  gh  = h32 + 16 * NN;              // 16*1536
    half_t* h16 = (half_t*)(gh + 16 * G2);    // 16*512
    const int tid = threadIdx.x, wave = tid >> 5, lane = tid & 31;
    const int bb0 = blockIdx.z * 16;
    for (int i = tid; i < 16 * NN; i += 256) { h32[i] = 0.f; h16[i] = (half_t)0.f; }
    __syncthreads();

    for (int t = 0; t < TT; ++t) {
        for (int tile = wave; tile < G2 / 16; tile += 8) {
            v8f c = {};
            const half_t* Bt = Whht + tile * 16 * NN;
            for (int kc = 0; kc < NN; kc += 32) {
                v16h a = frag_a(h16, NN, lane, kc);
                v16h b = frag_b(Bt, NN, lane, kc);
                c = __builtin_amdgcn_wmma_f32_16x16x32_f16(false, a, false, b,
                                                           (short)0, c, false, false);
            }
            const int n = lane & 15, hf = lane >> 4;
#pragma unroll
            for (int r = 0; r < 8; ++r)
                gh[(r + 8 * hf) * G2 + tile * 16 + n] = c[r];
        }
        __syncthreads();
        for (int i = tid; i < 16 * NN; i += 256) {
            const int b = i >> 9, d = i & 511;
            const float* g = gi + ((long long)(bb0 + b) * TT + t) * G2;
            const float rr = sigmoidf_(g[d]      + gh[b * G2 + d]      + bhh[d]);
            const float zz = sigmoidf_(g[NN + d] + gh[b * G2 + NN + d] + bhh[NN + d]);
            const float nn = tanhf(g[2 * NN + d] + rr * (gh[b * G2 + 2 * NN + d] + bhh[2 * NN + d]));
            const float h = (1.f - zz) * nn + zz * h32[i];
            h32[i] = h; h16[i] = (half_t)h;
            const long long bb = bb0 + b;
            out[((bb * 3 + 2) * (long long)NN + d) * TT + t] = h;   // channel 2
        }
        __syncthreads();
    }
}

// ---- small data-movement / conversion kernels ---------------------------
__global__ void k_zero16(half_t* p, long long n) {
    long long i = (long long)blockIdx.x * blockDim.x + threadIdx.x;
    if (i < n) p[i] = (half_t)0.f;
}
__global__ void k_cvt_direct(const float* s, half_t* d, int rows, int cols, int colsPad) {
    long long i = (long long)blockIdx.x * blockDim.x + threadIdx.x;
    if (i >= (long long)rows * colsPad) return;
    int r = (int)(i / colsPad), c = (int)(i % colsPad);
    d[i] = (half_t)(c < cols ? s[(long long)r * cols + c] : 0.f);
}
__global__ void k_cvt_trans(const float* s, half_t* d, int K, int Nc, int Kpad) {
    long long i = (long long)blockIdx.x * blockDim.x + threadIdx.x;
    if (i >= (long long)Nc * Kpad) return;
    int n = (int)(i / Kpad), k = (int)(i % Kpad);
    d[i] = (half_t)(k < K ? s[(long long)k * Nc + n] : 0.f);
}
__global__ void k_pack_sin(const float* x, const float* emb_s, half_t* d) {
    long long i = (long long)blockIdx.x * blockDim.x + threadIdx.x;
    if (i >= (long long)BB * NN * F1) return;
    int f = (int)(i % F1);
    int n = (int)((i / F1) % NN);
    int b = (int)(i / ((long long)NN * F1));
    float v = (f < TT) ? x[((long long)b * TT + f) * NN + n] : emb_s[n * 16 + (f - TT)];
    d[i] = (half_t)v;
}
__global__ void k_pack_tin(const float* x, const float* xm, const float* emb_t, half_t* d) {
    long long i = (long long)blockIdx.x * blockDim.x + threadIdx.x;
    if (i >= (long long)BB * TT * F2P) return;
    int f = (int)(i % F2P);
    int t = (int)((i / F2P) % TT);
    int b = (int)(i / ((long long)TT * F2P));
    float v = 0.f;
    if (f < NN)            v = x[((long long)b * TT + t) * NN + f];
    else if (f < NN + 4)   v = xm[((long long)b * TT + t) * 4 + (f - NN)];
    else if (f < NN + 20)  v = emb_t[t * 16 + (f - NN - 4)];
    d[i] = (half_t)v;
}
__global__ void k_copy_xnt(const float* x, float* out) {
    long long i = (long long)blockIdx.x * blockDim.x + threadIdx.x;
    if (i >= (long long)BB * NN * TT) return;
    int t = (int)(i % TT);
    int n = (int)((i / TT) % NN);
    int b = (int)(i / ((long long)NN * TT));
    out[((long long)b * 3 * NN + n) * TT + t] = x[((long long)b * TT + t) * NN + n];
}

// ---- host side ----------------------------------------------------------
static inline void launch_gemm(hipStream_t st, GemmP p, int batch) {
    const int ntiles = p.N / 16;
    dim3 g((unsigned)((ntiles + 15) / 16), (unsigned)(p.M / 16), (unsigned)batch);
    k_gemm<<<g, dim3(256), 0, st>>>(p);
}
static inline unsigned gsz(long long n) { return (unsigned)((n + 255) / 256); }

extern "C" void kernel_launch(void* const* d_in, const int* in_sizes, int n_in,
                              void* d_out, int out_size, void* d_ws, size_t ws_size,
                              hipStream_t stream) {
    (void)in_sizes; (void)n_in; (void)out_size; (void)ws_size;
    const float* x     = (const float*)d_in[0];
    const float* xm    = (const float*)d_in[1];
    const float* emb_s = (const float*)d_in[2];
    const float* Ws1   = (const float*)d_in[3];
    const float* bs1   = (const float*)d_in[4];
    const float* Ws2   = (const float*)d_in[5];
    const float* bs2   = (const float*)d_in[6];
    const float* emb_t = (const float*)d_in[7];
    const float* Wih1  = (const float*)d_in[8];
    const float* Whh1  = (const float*)d_in[9];
    const float* bih1  = (const float*)d_in[10];
    const float* bhh1  = (const float*)d_in[11];
    const float* Wg    = (const float*)d_in[12];
    const float* bg    = (const float*)d_in[13];
    const float* Wr1   = (const float*)d_in[14];
    const float* br1   = (const float*)d_in[15];
    const float* Wr2   = (const float*)d_in[16];
    const float* br2   = (const float*)d_in[17];
    const float* Wih2  = (const float*)d_in[18];
    const float* Whh2  = (const float*)d_in[19];
    const float* bih2  = (const float*)d_in[20];
    const float* bhh2  = (const float*)d_in[21];
    float* out = (float*)d_out;

    char* ws = (char*)d_ws;
    size_t off = 0;
    auto a16 = [&](size_t e) { half_t* p = (half_t*)(ws + off); off += ((e * 2 + 255) / 256) * 256; return p; };

    // f16 weights (B^T layouts)
    half_t* Ws1t  = a16((size_t)DD * F1);
    half_t* Ws2t  = a16((size_t)DD * DD);
    half_t* Wih1t = a16((size_t)G1 * F2P);
    half_t* Whh1t = a16((size_t)G1 * DD);
    half_t* Wg0t  = a16((size_t)DD * DD);
    half_t* Wg1t  = a16((size_t)DD * DD);
    half_t* Wr1t  = a16((size_t)TT * DD);
    half_t* Wr2t  = a16((size_t)TT * TP);
    half_t* Wih2t = a16((size_t)G2 * DD);
    half_t* Whh2t = a16((size_t)G2 * NN);
    // activations
    half_t* Xall  = a16((size_t)BB * MM * DD);   // Xs rows 0..511, Xt rows 512..847
    half_t* XT0   = a16((size_t)BB * DD * MMP);  // Xall^T, K-padded
    half_t* adjh  = a16((size_t)BB * MM * MMP);  // symmetric adjacency, K-padded
    half_t* tmph  = a16((size_t)BB * MM * DD);
    half_t* Hh1   = a16((size_t)BB * MM * DD);
    half_t* HT1   = a16((size_t)BB * DD * MMP);
    half_t* Hh2   = a16((size_t)BB * MM * DD);
    half_t* Qh    = a16((size_t)BB * NN * TP);
    // overlapped scratch region (gi2 reuses s_in/P1/t_in/gi1 space, all dead by then)
    size_t offU = off;
    half_t* s_in = (half_t*)(ws + offU);
    half_t* P1   = s_in + (size_t)BB * NN * F1;
    half_t* t_in = P1 + (size_t)BB * NN * DD;
    float*  gi1  = (float*)(t_in + (size_t)BB * TT * F2P);
    float*  gi2  = (float*)(ws + offU);

    // 1) weight conversion/transposition to f16
    k_cvt_trans<<<gsz((long long)DD * F1), 256, 0, stream>>>(Ws1, Ws1t, F1, DD, F1);
    k_cvt_trans<<<gsz((long long)DD * DD), 256, 0, stream>>>(Ws2, Ws2t, DD, DD, DD);
    k_cvt_direct<<<gsz((long long)G1 * F2P), 256, 0, stream>>>(Wih1, Wih1t, G1, 532, F2P);
    k_cvt_direct<<<gsz((long long)G1 * DD), 256, 0, stream>>>(Whh1, Whh1t, G1, DD, DD);
    k_cvt_trans<<<gsz((long long)DD * DD), 256, 0, stream>>>(Wg, Wg0t, DD, DD, DD);
    k_cvt_trans<<<gsz((long long)DD * DD), 256, 0, stream>>>(Wg + DD * DD, Wg1t, DD, DD, DD);
    k_cvt_trans<<<gsz((long long)TT * DD), 256, 0, stream>>>(Wr1, Wr1t, DD, TT, DD);
    k_cvt_trans<<<gsz((long long)TT * TP), 256, 0, stream>>>(Wr2, Wr2t, TT, TT, TP);
    k_cvt_direct<<<gsz((long long)G2 * DD), 256, 0, stream>>>(Wih2, Wih2t, G2, DD, DD);
    k_cvt_direct<<<gsz((long long)G2 * NN), 256, 0, stream>>>(Whh2, Whh2t, G2, NN, NN);

    // 2) input packing, 3) zero pad-carrying buffers
    k_pack_sin<<<gsz((long long)BB * NN * F1), 256, 0, stream>>>(x, emb_s, s_in);
    k_pack_tin<<<gsz((long long)BB * TT * F2P), 256, 0, stream>>>(x, xm, emb_t, t_in);
    k_zero16<<<gsz((long long)BB * MM * MMP), 256, 0, stream>>>(adjh, (long long)BB * MM * MMP);
    k_zero16<<<gsz((long long)BB * DD * MMP), 256, 0, stream>>>(XT0, (long long)BB * DD * MMP);
    k_zero16<<<gsz((long long)BB * DD * MMP), 256, 0, stream>>>(HT1, (long long)BB * DD * MMP);
    k_zero16<<<gsz((long long)BB * NN * TP), 256, 0, stream>>>(Qh, (long long)BB * NN * TP);

    GemmP p;
    // 4) P1 = elu(s_in @ Ws1 + bs1)      (M = B*N fused, batch=1)
    p = {s_in, 0, F1, Ws1t, 0, F1, bs1, nullptr, 0, 0, P1, 0, DD, nullptr, 0, 0,
         BB * NN, DD, F1, 1, 0};
    launch_gemm(stream, p, 1);
    // 5) Xs = P1 @ Ws2 + bs2  -> Xall rows [0,512), XT0 cols [0,512)
    p = {P1, (long long)NN * DD, DD, Ws2t, 0, DD, bs2, nullptr, 0, 0,
         Xall, (long long)MM * DD, DD, XT0, (long long)DD * MMP, MMP,
         NN, DD, DD, 0, 0};
    launch_gemm(stream, p, BB);
    // 6) gi1 = t_in @ Wih1^T + bih1      (M = B*T fused)
    p = {t_in, 0, F2P, Wih1t, 0, F2P, bih1, gi1, 0, G1, nullptr, 0, 0, nullptr, 0, 0,
         BB * TT, G1, F2P, 0, 0};
    launch_gemm(stream, p, 1);
    // 7) GRU1 scan -> Xall rows [512,848), XT0 cols [512,848)
    k_gru1<<<dim3(1, 1, BB / 16), dim3(256),
             (16 * DD * 4 + 16 * G1 * 4 + 16 * DD * 2), stream>>>(gi1, Whh1t, bhh1, Xall, XT0);
    // 8) adj = tanh(relu(Xall @ Xall^T)) + eps*I   (symmetric)
    p = {Xall, (long long)MM * DD, DD, Xall, (long long)MM * DD, DD, nullptr,
         nullptr, 0, 0, adjh, (long long)MM * MMP, MMP, nullptr, 0, 0,
         MM, MM, DD, 2, 1};
    launch_gemm(stream, p, BB);
    // 9) GCN layer 1: tmp = adj @ Xall ; H1 = elu(tmp @ Wg0 + bg0)
    p = {adjh, (long long)MM * MMP, MMP, XT0, (long long)DD * MMP, MMP, nullptr,
         nullptr, 0, 0, tmph, (long long)MM * DD, DD, nullptr, 0, 0,
         MM, DD, MMP, 0, 0};
    launch_gemm(stream, p, BB);
    p = {tmph, (long long)MM * DD, DD, Wg0t, 0, DD, bg, nullptr, 0, 0,
         Hh1, (long long)MM * DD, DD, HT1, (long long)DD * MMP, MMP,
         MM, DD, DD, 1, 0};
    launch_gemm(stream, p, BB);
    // 10) GCN layer 2
    p = {adjh, (long long)MM * MMP, MMP, HT1, (long long)DD * MMP, MMP, nullptr,
         nullptr, 0, 0, tmph, (long long)MM * DD, DD, nullptr, 0, 0,
         MM, DD, MMP, 0, 0};
    launch_gemm(stream, p, BB);
    p = {tmph, (long long)MM * DD, DD, Wg1t, 0, DD, bg + DD, nullptr, 0, 0,
         Hh2, (long long)MM * DD, DD, nullptr, 0, 0,
         MM, DD, DD, 1, 0};
    launch_gemm(stream, p, BB);
    // 11) Q = elu(Hs @ Wr1 + br1)
    p = {Hh2, (long long)MM * DD, DD, Wr1t, 0, DD, br1, nullptr, 0, 0,
         Qh, (long long)NN * TP, TP, nullptr, 0, 0,
         NN, TT, DD, 1, 0};
    launch_gemm(stream, p, BB);
    // 12) rs = Q @ Wr2 + br2 -> out channel 1
    p = {Qh, (long long)NN * TP, TP, Wr2t, 0, TP, br2,
         out + (long long)NN * TT, (long long)3 * NN * TT, TT,
         nullptr, 0, 0, nullptr, 0, 0,
         NN, TT, TP, 0, 0};
    launch_gemm(stream, p, BB);
    // 13) gi2 = Ht @ Wih2^T + bih2
    p = {Hh2 + (size_t)NN * DD, (long long)MM * DD, DD, Wih2t, 0, DD, bih2,
         gi2, (long long)TT * G2, G2, nullptr, 0, 0, nullptr, 0, 0,
         TT, G2, DD, 0, 0};
    launch_gemm(stream, p, BB);
    // 14) GRU2 scan -> out channel 2 (transposed)
    k_gru2<<<dim3(1, 1, BB / 16), dim3(256),
             (16 * NN * 4 + 16 * G2 * 4 + 16 * NN * 2), stream>>>(gi2, Whh2t, bhh2, out);
    // 15) out channel 0 = x transposed
    k_copy_xnt<<<gsz((long long)BB * NN * TT), 256, 0, stream>>>(x, out);
}
